// GraphMLPMixer_82094004896370
// MI455X (gfx1250) — matrix operand order repaired and checked by reading, not compiled, not run on hardware
//
#include <hip/hip_runtime.h>
#include <hip/hip_bf16.h>

// ---------------- constants (match reference) ----------------
#define Bg   256
#define Pp   32
#define Dd   128
#define BP   8192          // B*P
#define Nn   200000
#define NSs  500000
#define ESs  1000000
#define RW_  20
#define PRW_ 8
#define TH_  64
#define CH_  512

typedef __attribute__((ext_vector_type(16))) _Float16 v16h;
typedef __attribute__((ext_vector_type(8)))  float    v8f;

union HPack { v16h v; unsigned u[8]; };

__device__ __forceinline__ float gelu_f(float x) {
  float x3 = x * x * x;
  return 0.5f * x * (1.0f + tanhf(0.7978845608028654f * (x + 0.044715f * x3)));
}

__device__ __forceinline__ float wsum(float v) {
#pragma unroll
  for (int m = 16; m > 0; m >>= 1) v += __shfl_xor(v, m, 32);
  return v;
}

// ---------------- weight repack: f32 row-major [KD,ND] -> f16 WMMA-B fragments ----
// Fragment order: (ct, kc, lane, vv): lane L holds B[k..k+1, col] pairs with
// k = kc*32 + (L>>4)*16 + 2*vv, col = ct*16 + (L&15). 8 u32 words per lane,
// so each lane's fragment = 2 contiguous b128 loads at GEMM time.
__global__ void k_repack(const float* __restrict__ W, unsigned* __restrict__ Wf,
                         int KD, int ND) {
  int gid = blockIdx.x * blockDim.x + threadIdx.x;
  int total = (KD * ND) >> 1;
  if (gid >= total) return;
  int vv = gid & 7;
  int L  = (gid >> 3) & 31;
  int frag = gid >> 8;                 // ct*(KD/32)+kc
  int KC = KD >> 5;
  int kc = frag % KC;
  int ct = frag / KC;
  int k   = kc * 32 + ((L >> 4) << 4) + (vv << 1);
  int col = ct * 16 + (L & 15);
  union { _Float16 h[2]; unsigned u; } p;
  p.h[0] = (_Float16)W[(long)k * ND + col];
  p.h[1] = (_Float16)W[(long)(k + 1) * ND + col];
  Wf[gid] = p.u;
}

// ---------------- WMMA GEMM: out[M,ND] = res? + act(in1(+in2)[M,KD] @ W[KD,ND] + bias) ----
// grid.x = M/32, 256 threads (8 waves). Wave w owns column tiles w + 8q.
// Both 16-row tiles kept in flight so each B fragment is loaded once.
template<int KD, int ND, int ACT, int ADD2, int RES>
__global__ __launch_bounds__(256) void k_gemm(
    const float* in1, const float* in2,
    const unsigned* __restrict__ Wf, const float* __restrict__ bias,
    const float* res, float* out)
{
  __shared__ _Float16 lds[32 * KD];
  const int  tid     = threadIdx.x;
  const long rowBase = (long)blockIdx.x * 32;

  // stage (in1[+in2]) rows -> LDS as f16
  for (int idx = tid; idx < 32 * KD; idx += 256) {
    int  r  = idx / KD, c = idx - r * KD;
    long gi = (rowBase + r) * KD + c;
    float v = in1[gi];
    if (ADD2) v += in2[gi];
    lds[idx] = (_Float16)v;
  }
  __syncthreads();

  const int w      = tid >> 5;
  const int lane   = tid & 31;
  const int lane16 = lane & 15;
  const int khalf  = lane >> 4;
  const unsigned* ldsu = (const unsigned*)lds;
  constexpr int KC = KD / 32;

  for (int q = 0; q < ND / 128; ++q) {
    const int ct  = w + q * 8;
    const int col = ct * 16 + lane16;
    v8f acc0 = {}, acc1 = {};
    for (int kc = 0; kc < KC; ++kc) {
      HPack b;
      const uint4* bp = (const uint4*)(Wf + (((ct * KC + kc) * 32 + lane) << 3));
      uint4 b0 = bp[0], b1 = bp[1];
      b.u[0] = b0.x; b.u[1] = b0.y; b.u[2] = b0.z; b.u[3] = b0.w;
      b.u[4] = b1.x; b.u[5] = b1.y; b.u[6] = b1.z; b.u[7] = b1.w;
      HPack a;
      const int abase0 = lane16 * KD + kc * 32;
#pragma unroll
      for (int vv = 0; vv < 8; ++vv) {          // A: 16-bit 16x32 layout
        const int k0 = khalf * 16 + vv * 2;
        a.u[vv] = ldsu[(abase0 + k0) >> 1];
      }
      acc0 = __builtin_amdgcn_wmma_f32_16x16x32_f16(
          false, a.v, false, b.v, (short)0, acc0, false, false);
      const int abase1 = (16 + lane16) * KD + kc * 32;
#pragma unroll
      for (int vv = 0; vv < 8; ++vv) {
        const int k0 = khalf * 16 + vv * 2;
        a.u[vv] = ldsu[(abase1 + k0) >> 1];
      }
      acc1 = __builtin_amdgcn_wmma_f32_16x16x32_f16(
          false, a.v, false, b.v, (short)0, acc1, false, false);
    }
#pragma unroll
    for (int j = 0; j < 8; ++j) {               // C: VGPR j -> M=j (lo half) / 8+j (hi)
      const int mrow = khalf ? 8 + j : j;
      long row = rowBase + mrow;
      float v = acc0[j] + bias[col];
      if (ACT == 1) v = fmaxf(v, 0.0f);
      if (ACT == 2) v = gelu_f(v);
      if (RES) v += res[row * ND + col];
      out[row * ND + col] = v;
      row = rowBase + 16 + mrow;
      float v1 = acc1[j] + bias[col];
      if (ACT == 1) v1 = fmaxf(v1, 0.0f);
      if (ACT == 2) v1 = gelu_f(v1);
      if (RES) v1 += res[row * ND + col];
      out[row * ND + col] = v1;
    }
  }
}

// ---------------- elementwise / scatter kernels ----------------
__global__ void k_node_enc(const int* __restrict__ x_atom, const float* __restrict__ rw,
                           const float* __restrict__ aemb, const float* __restrict__ rwW,
                           const float* __restrict__ rwB, float* __restrict__ out, int n) {
  int gid = blockIdx.x * blockDim.x + threadIdx.x;
  if (gid >= n * Dd) return;
  int node = gid >> 7, d = gid & 127;
  float s = aemb[x_atom[node] * Dd + d] + rwB[d];
#pragma unroll
  for (int k = 0; k < RW_; ++k) s += rw[node * RW_ + k] * rwW[k * Dd + d];
  out[gid] = s;
}

__global__ void k_count(const int* __restrict__ map, float* __restrict__ cnt, int n) {
  int i = blockIdx.x * blockDim.x + threadIdx.x;
  if (i < n) atomicAdd(&cnt[map[i]], 1.0f);
}

__global__ void k_gather(const float* __restrict__ src, const int* __restrict__ map,
                         float* __restrict__ dst, int rows) {
  long gid = (long)blockIdx.x * blockDim.x + threadIdx.x;
  if (gid >= (long)rows * 32) return;
  int i = (int)(gid >> 5), c = ((int)gid & 31) * 4;
  float4 v = *(const float4*)(src + (long)map[i] * Dd + c);
  *(float4*)(dst + (long)i * Dd + c) = v;
}

__global__ void k_gather_mean(const float* __restrict__ src, const int* __restrict__ map,
                              const float* __restrict__ cnt, float* __restrict__ dst, int rows) {
  long gid = (long)blockIdx.x * blockDim.x + threadIdx.x;
  if (gid >= (long)rows * 32) return;
  int i = (int)(gid >> 5), c = ((int)gid & 31) * 4;
  int n = map[i];
  float inv = 1.0f / fmaxf(cnt[n], 1.0f);
  float4 v = *(const float4*)(src + (long)n * Dd + c);
  v.x *= inv; v.y *= inv; v.z *= inv; v.w *= inv;
  *(float4*)(dst + (long)i * Dd + c) = v;
}

__global__ void k_add_gather(const float* __restrict__ t, const int* __restrict__ map,
                             float* __restrict__ x, int rows) {
  long gid = (long)blockIdx.x * blockDim.x + threadIdx.x;
  if (gid >= (long)rows * 32) return;
  int i = (int)(gid >> 5), c = ((int)gid & 31) * 4;
  float4 a = *(const float4*)(x + (long)i * Dd + c);
  float4 b = *(const float4*)(t + (long)map[i] * Dd + c);
  a.x += b.x; a.y += b.y; a.z += b.z; a.w += b.w;
  *(float4*)(x + (long)i * Dd + c) = a;
}

__global__ void k_scatter_add(const float* __restrict__ src, const int* __restrict__ map,
                              float* __restrict__ dst, int rows) {
  long gid = (long)blockIdx.x * blockDim.x + threadIdx.x;
  if (gid >= (long)rows * 32) return;
  int i = (int)(gid >> 5), c = ((int)gid & 31) * 4;
  int t = map[i];
  float4 v = *(const float4*)(src + (long)i * Dd + c);
  float* a = dst + (long)t * Dd + c;
  atomicAdd(a + 0, v.x); atomicAdd(a + 1, v.y);
  atomicAdd(a + 2, v.z); atomicAdd(a + 3, v.w);
}

__global__ void k_edge(const float* __restrict__ x, const int* __restrict__ esrc,
                       const int* __restrict__ edst, const int* __restrict__ emap,
                       const int* __restrict__ eattr, const float* __restrict__ bemb,
                       float* __restrict__ agg, int es) {
  long gid = (long)blockIdx.x * blockDim.x + threadIdx.x;
  if (gid >= (long)es * 32) return;
  int e = (int)(gid >> 5), c = ((int)gid & 31) * 4;
  int s = esrc[e], d = edst[e];
  int ea = eattr[emap[e]];
  float4 xv = *(const float4*)(x + (long)s * Dd + c);
  float4 bv = *(const float4*)(bemb + (long)ea * Dd + c);
  float4 m;
  m.x = fmaxf(xv.x + bv.x, 0.f); m.y = fmaxf(xv.y + bv.y, 0.f);
  m.z = fmaxf(xv.z + bv.z, 0.f); m.w = fmaxf(xv.w + bv.w, 0.f);
  float* a = agg + (long)d * Dd + c;
  atomicAdd(a + 0, m.x); atomicAdd(a + 1, m.y);
  atomicAdd(a + 2, m.z); atomicAdd(a + 3, m.w);
}

__global__ void k_divrow(float* __restrict__ buf, const float* __restrict__ cnt, int rows) {
  int gid = blockIdx.x * blockDim.x + threadIdx.x;
  if (gid >= rows * Dd) return;
  buf[gid] /= fmaxf(cnt[gid >> 7], 1.0f);
}

__global__ void k_minit(const float* __restrict__ psum, const float* __restrict__ cnt,
                        const float* __restrict__ pe, const float* __restrict__ prwW,
                        const float* __restrict__ prwB, float* __restrict__ m) {
  int gid = blockIdx.x * blockDim.x + threadIdx.x;
  if (gid >= BP * Dd) return;
  int j = gid >> 7, d = gid & 127;
  float s = psum[gid] / fmaxf(cnt[j], 1.0f) + prwB[d];
#pragma unroll
  for (int k = 0; k < PRW_; ++k) s += pe[j * PRW_ + k] * prwW[k * Dd + d];
  m[gid] = s;
}

__global__ __launch_bounds__(256) void k_ln(const float* __restrict__ in,
    const float* __restrict__ g, const float* __restrict__ bta,
    float* __restrict__ out, int rows) {
  int w = threadIdx.x >> 5, L = threadIdx.x & 31;
  int row = blockIdx.x * 8 + w;
  if (row >= rows) return;
  const float* r = in + (long)row * Dd;
  float v[4], s = 0.f;
#pragma unroll
  for (int j = 0; j < 4; ++j) { v[j] = r[L + 32 * j]; s += v[j]; }
  s = wsum(s);
  float mean = s * (1.f / 128.f);
  float q = 0.f;
#pragma unroll
  for (int j = 0; j < 4; ++j) { float d2 = v[j] - mean; q += d2 * d2; }
  q = wsum(q);
  float inv = rsqrtf(q * (1.f / 128.f) + 1e-5f);
  float* o = out + (long)row * Dd;
#pragma unroll
  for (int j = 0; j < 4; ++j) { int d = L + 32 * j; o[d] = (v[j] - mean) * inv * g[d] + bta[d]; }
}

// token mixing (per graph): m += (gelu(ln(m)^T @ tW1 + tb1) @ tW2 + tb2)^T
__global__ __launch_bounds__(256) void k_token(float* __restrict__ m,
    const float* __restrict__ g, const float* __restrict__ bta,
    const float* __restrict__ tW1, const float* __restrict__ tb1,
    const float* __restrict__ tW2, const float* __restrict__ tb2) {
  __shared__ float lnm[Pp][Dd];
  __shared__ float hh[Dd][TH_];
  const int b = blockIdx.x;
  const int w = threadIdx.x >> 5, L = threadIdx.x & 31;
  float* mb = m + (long)b * Pp * Dd;
  for (int pp = 0; pp < 4; ++pp) {
    const int p = w * 4 + pp;
    float v[4], s = 0.f;
#pragma unroll
    for (int j = 0; j < 4; ++j) { v[j] = mb[p * Dd + L + 32 * j]; s += v[j]; }
    s = wsum(s);
    const float mean = s * (1.f / 128.f);
    float q = 0.f;
#pragma unroll
    for (int j = 0; j < 4; ++j) { float d2 = v[j] - mean; q += d2 * d2; }
    q = wsum(q);
    const float inv = rsqrtf(q * (1.f / 128.f) + 1e-5f);
#pragma unroll
    for (int j = 0; j < 4; ++j) { int d = L + 32 * j; lnm[p][d] = (v[j] - mean) * inv * g[d] + bta[d]; }
  }
  __syncthreads();
  const int d  = threadIdx.x >> 1;
  const int t0 = (threadIdx.x & 1) * 32;
  for (int t = t0; t < t0 + 32; ++t) {
    float s = tb1[t];
#pragma unroll
    for (int p = 0; p < Pp; ++p) s += lnm[p][d] * tW1[p * TH_ + t];
    hh[d][t] = gelu_f(s);
  }
  __syncthreads();
  const int p0 = (threadIdx.x & 1) * 16;
  for (int p = p0; p < p0 + 16; ++p) {
    float s = tb2[p];
#pragma unroll
    for (int t = 0; t < TH_; ++t) s += hh[d][t] * tW2[t * Pp + p];
    mb[p * Dd + d] += s;
  }
}

__global__ void k_maskpool(const float* __restrict__ m, const unsigned char* __restrict__ mask,
                           float* __restrict__ out) {
  int gid = blockIdx.x * blockDim.x + threadIdx.x;
  if (gid >= Bg * Dd) return;
  int b = gid >> 7, d = gid & 127;
  float s = 0.f, c = 0.f;
  for (int p = 0; p < Pp; ++p) {
    float mm = mask[b * Pp + p] ? 1.f : 0.f;
    s += mm * m[(long)b * Pp * Dd + p * Dd + d];
    c += mm;
  }
  out[gid] = s / fmaxf(c, 1.f);
}

__global__ void k_final(const float* __restrict__ r1, const float* __restrict__ oW2,
                        const float* __restrict__ ob2, float* __restrict__ out) {
  int w = threadIdx.x >> 5, L = threadIdx.x & 31;
  int b = blockIdx.x * 8 + w;
  if (b >= Bg) return;
  float s = 0.f;
#pragma unroll
  for (int j = 0; j < 4; ++j) { int d = L + 32 * j; s += r1[(long)b * Dd + d] * oW2[d]; }
  s = wsum(s);
  if (L == 0) out[b] = s + ob2[0];
}

// ---------------- host ----------------
extern "C" void kernel_launch(void* const* d_in, const int* in_sizes, int n_in,
                              void* d_out, int out_size, void* d_ws, size_t ws_size,
                              hipStream_t stream) {
  const int*   x_atom   = (const int*)d_in[0];
  const float* rw_pos   = (const float*)d_in[1];
  const int*   edge_attr= (const int*)d_in[2];
  const int*   nmap     = (const int*)d_in[3];
  const int*   emap     = (const int*)d_in[4];
  const int*   eidx     = (const int*)d_in[5];
  const int*   batch_x  = (const int*)d_in[6];
  const float* patch_pe = (const float*)d_in[7];
  const unsigned char* mask = (const unsigned char*)d_in[8];
  const float* atom_emb = (const float*)d_in[9];
  const float* bond_emb = (const float*)d_in[10];
  const float* rw_W = (const float*)d_in[11];
  const float* rw_b = (const float*)d_in[12];
  const float* prw_W = (const float*)d_in[13];
  const float* prw_b = (const float*)d_in[14];
  const float* gnn_W = (const float*)d_in[15];
  const float* gnn_b = (const float*)d_in[16];
  const float* U_W = (const float*)d_in[17];
  const float* U_b = (const float*)d_in[18];
  const float* ln1_g = (const float*)d_in[19];
  const float* ln1_b = (const float*)d_in[20];
  const float* tW1 = (const float*)d_in[21];
  const float* tb1 = (const float*)d_in[22];
  const float* tW2 = (const float*)d_in[23];
  const float* tb2 = (const float*)d_in[24];
  const float* ln2_g = (const float*)d_in[25];
  const float* ln2_b = (const float*)d_in[26];
  const float* cW1 = (const float*)d_in[27];
  const float* cb1 = (const float*)d_in[28];
  const float* cW2 = (const float*)d_in[29];
  const float* cb2 = (const float*)d_in[30];
  const float* oW1 = (const float*)d_in[31];
  const float* ob1 = (const float*)d_in[32];
  const float* oW2 = (const float*)d_in[33];
  const float* ob2 = (const float*)d_in[34];
  float* out = (float*)d_out;

  char* ws = (char*)d_ws;
  // workspace layout (bytes); aggbuf region aliased by nodebuf (GNN pre-steps) & hbuf (mixer)
  const size_t offX  = 0;                              // NS*128 f32
  const size_t offA  = 256000000ull;                   // NS*128 f32
  const size_t offPB = 512000000ull;                   // 8192*128 f32
  const size_t offPB2= offPB  + 4194304ull;
  const size_t offM  = offPB2 + 4194304ull;
  const size_t offCP = offM   + 4194304ull;            // 8192 f32
  const size_t offCN = offCP  + 32768ull;              // N f32
  const size_t offWF = offCN  + 800000ull;             // repacked f16 weights (u32 words)
  float* xbuf   = (float*)(ws + offX);
  float* aggbuf = (float*)(ws + offA);
  float* nodebuf= aggbuf;            // N*128, live only outside conv
  float* hbuf   = aggbuf;            // 8192*512, live only in mixer
  float* pbuf   = (float*)(ws + offPB);
  float* pbuf2  = (float*)(ws + offPB2);
  float* mbuf   = (float*)(ws + offM);
  float* cntP   = (float*)(ws + offCP);
  float* cntN   = (float*)(ws + offCN);
  unsigned* wfBase = (unsigned*)(ws + offWF);
  unsigned* wfGNN = wfBase;                      // 4 * 8192 words
  unsigned* wfU   = wfBase + 4 * 8192;           // 3 * 8192 words
  unsigned* wfC1  = wfBase + 7 * 8192;           // 2 * 32768 words
  unsigned* wfC2  = wfBase + 7 * 8192 + 2 * 32768;
  unsigned* wfO1  = wfBase + 7 * 8192 + 4 * 32768;

  const int blk = 256;
  const int gNS32 = (int)(((long)NSs * 32 + blk - 1) / blk);
  const int gES32 = (int)(((long)ESs * 32 + blk - 1) / blk);

  // ---- repack all weight matrices to f16 WMMA fragments (once per launch) ----
  for (int i = 0; i < 4; ++i)
    k_repack<<<32, blk, 0, stream>>>(gnn_W + (size_t)i * 16384, wfGNN + i * 8192, 128, 128);
  for (int i = 0; i < 3; ++i)
    k_repack<<<32, blk, 0, stream>>>(U_W + (size_t)i * 16384, wfU + i * 8192, 128, 128);
  for (int l = 0; l < 2; ++l) {
    k_repack<<<128, blk, 0, stream>>>(cW1 + (size_t)l * 65536, wfC1 + l * 32768, 128, 512);
    k_repack<<<128, blk, 0, stream>>>(cW2 + (size_t)l * 65536, wfC2 + l * 32768, 512, 128);
  }
  k_repack<<<32, blk, 0, stream>>>(oW1, wfO1, 128, 128);

  // segment counts (constant for the whole launch)
  hipMemsetAsync(cntP, 0, (size_t)BP * 4, stream);
  hipMemsetAsync(cntN, 0, (size_t)Nn * 4, stream);
  k_count<<<(NSs + blk - 1) / blk, blk, 0, stream>>>(batch_x, cntP, NSs);
  k_count<<<(NSs + blk - 1) / blk, blk, 0, stream>>>(nmap, cntN, NSs);

  // node encoder -> nodebuf, gather -> xbuf
  k_node_enc<<<(Nn * Dd + blk - 1) / blk, blk, 0, stream>>>(x_atom, rw_pos, atom_emb, rw_W, rw_b, nodebuf, Nn);
  k_gather<<<gNS32, blk, 0, stream>>>(nodebuf, nmap, xbuf, NSs);

  for (int i = 0; i < 4; ++i) {
    if (i > 0) {
      // sub = seg_mean(x, batch_x); x += relu(sub @ U_W + U_b); x = seg_mean(x, nmap)[nmap]
      hipMemsetAsync(pbuf, 0, (size_t)BP * Dd * 4, stream);
      k_scatter_add<<<gNS32, blk, 0, stream>>>(xbuf, batch_x, pbuf, NSs);
      k_divrow<<<(BP * Dd) / blk, blk, 0, stream>>>(pbuf, cntP, BP);
      k_gemm<128, 128, 1, 0, 0><<<BP / 32, blk, 0, stream>>>(
          pbuf, (const float*)nullptr, wfU + (i - 1) * 8192, U_b + (i - 1) * 128,
          (const float*)nullptr, pbuf2);
      k_add_gather<<<gNS32, blk, 0, stream>>>(pbuf2, batch_x, xbuf, NSs);
      hipMemsetAsync(nodebuf, 0, (size_t)Nn * Dd * 4, stream);
      k_scatter_add<<<gNS32, blk, 0, stream>>>(xbuf, nmap, nodebuf, NSs);
      k_gather_mean<<<gNS32, blk, 0, stream>>>(nodebuf, nmap, cntN, xbuf, NSs);
    }
    // GINE conv
    hipMemsetAsync(aggbuf, 0, (size_t)NSs * Dd * 4, stream);
    k_edge<<<gES32, blk, 0, stream>>>(xbuf, eidx, eidx + ESs, emap, edge_attr, bond_emb, aggbuf, ESs);
    k_gemm<128, 128, 1, 1, 1><<<NSs / 32, blk, 0, stream>>>(
        xbuf, aggbuf, wfGNN + i * 8192, gnn_b + i * 128, xbuf, xbuf);
  }

  // patch readout: m = seg_mean(x, batch_x) + patch_pe @ prw_W + prw_b
  hipMemsetAsync(pbuf, 0, (size_t)BP * Dd * 4, stream);
  k_scatter_add<<<gNS32, blk, 0, stream>>>(xbuf, batch_x, pbuf, NSs);
  k_minit<<<(BP * Dd) / blk, blk, 0, stream>>>(pbuf, cntP, patch_pe, prw_W, prw_b, mbuf);

  // mixer
  for (int l = 0; l < 2; ++l) {
    k_token<<<Bg, blk, 0, stream>>>(mbuf, ln1_g + l * 128, ln1_b + l * 128,
                                    tW1 + l * Pp * TH_, tb1 + l * TH_,
                                    tW2 + l * TH_ * Pp, tb2 + l * Pp);
    k_ln<<<BP / 8, blk, 0, stream>>>(mbuf, ln2_g + l * 128, ln2_b + l * 128, pbuf2, BP);
    k_gemm<128, 512, 2, 0, 0><<<BP / 32, blk, 0, stream>>>(
        pbuf2, (const float*)nullptr, wfC1 + l * 32768, cb1 + l * 512,
        (const float*)nullptr, hbuf);
    k_gemm<512, 128, 0, 0, 1><<<BP / 32, blk, 0, stream>>>(
        hbuf, (const float*)nullptr, wfC2 + l * 32768, cb2 + l * 128, mbuf, mbuf);
  }

  // head
  k_maskpool<<<(Bg * Dd) / blk, blk, 0, stream>>>(mbuf, mask, pbuf2);
  k_gemm<128, 128, 1, 0, 0><<<Bg / 32, blk, 0, stream>>>(
      pbuf2, (const float*)nullptr, wfO1, ob1, (const float*)nullptr, pbuf);
  k_final<<<Bg / 8, blk, 0, stream>>>(pbuf, oW2, ob2, out);
}
